// FuseMamba_43344809951446
// MI455X (gfx1250) — matrix operand order repaired and checked by reading, not compiled, not run on hardware
//
#include <hip/hip_runtime.h>
#include <hip/hip_bf16.h>
#include <math.h>

// ---------------------------------------------------------------------------
// CDNA5 / gfx1250 fused Mamba block. wave32, WMMA bf16 16x16x32, f32 accum.
// B operands staged TRANSPOSED in LDS so fragments load as ds_load_b128.
// ---------------------------------------------------------------------------

typedef __bf16 bf16_t;
typedef __attribute__((ext_vector_type(16))) __bf16 v16bf;
typedef __attribute__((ext_vector_type(8)))  __bf16 v8bf;
typedef __attribute__((ext_vector_type(4)))  __bf16 v4bf;
typedef __attribute__((ext_vector_type(8)))  float  v8f;

#define BATCH   256
#define NTENS_ELEMS (256 * 64 * 64)   // 1048576 per tensor

static __device__ __forceinline__ v8f wmma_bf16(v16bf a, v16bf b, v8f c) {
  return __builtin_amdgcn_wmma_f32_16x16x32_bf16(
      false, a, false, b, (short)0, c, false, false);
}

static __device__ __forceinline__ v16bf cat8(v8bf lo, v8bf hi) {
  return __builtin_shufflevector(lo, hi, 0, 1, 2, 3, 4, 5, 6, 7,
                                 8, 9, 10, 11, 12, 13, 14, 15);
}

// A-fragment (16x32 bf16) from row-major LDS (stride S elements).
// Lane l: M = m0+(l&15); K = k0+16*(l>>4)+e
// Both halves are 8 contiguous bf16 (16B, aligned) -> ds_load_b128 x2.
static __device__ __forceinline__ v16bf ldsA(const bf16_t* base, int S,
                                             int m0, int k0, int lane) {
  const bf16_t* p = base + (m0 + (lane & 15)) * S + k0 + (lane >> 4) * 16;
  v8bf lo = *(const v8bf*)(p);
  v8bf hi = *(const v8bf*)(p + 8);
  return cat8(lo, hi);
}

// B-fragment (32x16 bf16) from a TRANSPOSED LDS matrix Bt[n][k] (stride S).
// Lane l: N = n0+(l&15); K = k0+16*(l>>4)+e -> 16 contiguous bf16.
static __device__ __forceinline__ v16bf ldsBT(const bf16_t* baseT, int S,
                                              int k0, int n0, int lane) {
  const bf16_t* p = baseT + (n0 + (lane & 15)) * S + k0 + (lane >> 4) * 16;
  v8bf lo = *(const v8bf*)(p);
  v8bf hi = *(const v8bf*)(p + 8);
  return cat8(lo, hi);
}

// B-fragment from GLOBAL row-major (N,K) f32 weight: B[k][n] = W[n*K + k].
static __device__ __forceinline__ v16bf gldB(const float* W, int K,
                                             int n0, int k0, int lane, int Nvalid) {
  int n = n0 + (lane & 15);
  int kk = k0 + (lane >> 4) * 16;
  v16bf b;
  if (n < Nvalid) {
    const float4* p = (const float4*)(W + n * K + kk);
#pragma unroll
    for (int q = 0; q < 4; ++q) {
      float4 v = p[q];
      b[q * 4 + 0] = (bf16_t)v.x; b[q * 4 + 1] = (bf16_t)v.y;
      b[q * 4 + 2] = (bf16_t)v.z; b[q * 4 + 3] = (bf16_t)v.w;
    }
  } else {
#pragma unroll
    for (int e = 0; e < 16; ++e) b[e] = (bf16_t)0.f;
  }
  return b;
}

// A-fragment from GLOBAL row-major (M,K) f32 weight.
static __device__ __forceinline__ v16bf gldA(const float* W, int K,
                                             int m0, int k0, int lane) {
  const float* p = W + (m0 + (lane & 15)) * K + k0 + (lane >> 4) * 16;
  v16bf a;
#pragma unroll
  for (int q = 0; q < 2; ++q) {
    float4 v = ((const float4*)p)[q];
    a[q * 4 + 0] = (bf16_t)v.x; a[q * 4 + 1] = (bf16_t)v.y;
    a[q * 4 + 2] = (bf16_t)v.z; a[q * 4 + 3] = (bf16_t)v.w;
  }
#pragma unroll
  for (int q = 0; q < 2; ++q) {
    float4 v = ((const float4*)(p + 8))[q];
    a[8 + q * 4 + 0] = (bf16_t)v.x; a[8 + q * 4 + 1] = (bf16_t)v.y;
    a[8 + q * 4 + 2] = (bf16_t)v.z; a[8 + q * 4 + 3] = (bf16_t)v.w;
  }
  return a;
}

static __device__ __forceinline__ float sigmoidf_(float x) {
  return 1.f / (1.f + __expf(-x));
}
static __device__ __forceinline__ float softplusf_(float x) {
  return (x > 20.f) ? x : log1pf(__expf(x));
}
static __device__ __forceinline__ v8f vzero8() {
  v8f z = {0.f, 0.f, 0.f, 0.f, 0.f, 0.f, 0.f, 0.f};
  return z;
}

// Stage a (64x64) f32 global tile TRANSPOSED into LDS bf16: dst[n*64+k]=src[k*64+n]
template <int NTHREADS>
static __device__ __forceinline__ void stageT(const float* __restrict__ src,
                                              bf16_t* __restrict__ dstT, int tid) {
  for (int i4 = tid; i4 < 1024; i4 += NTHREADS) {
    float4 v = ((const float4*)src)[i4];
    int elem = i4 * 4;
    int k = elem >> 6, n = elem & 63;
    dstT[(n + 0) * 64 + k] = (bf16_t)v.x;
    dstT[(n + 1) * 64 + k] = (bf16_t)v.y;
    dstT[(n + 2) * 64 + k] = (bf16_t)v.z;
    dstT[(n + 3) * 64 + k] = (bf16_t)v.w;
  }
}

// ---------------------------------------------------------------------------
// Kernel 1: c1 = W*x1+b ; a1=c1 ; a2=c1*(W*x2+b) ; a3=c1*(W*x3+b)
// ---------------------------------------------------------------------------
__global__ __launch_bounds__(128) void k_front_conv(
    const float* __restrict__ x1, const float* __restrict__ x2,
    const float* __restrict__ x3, const float* __restrict__ W,
    const float* __restrict__ cb,
    float* __restrict__ A1, float* __restrict__ A2, float* __restrict__ A3) {
  __shared__ bf16_t s1[64 * 64];  // transposed activations
  __shared__ bf16_t s2[64 * 64];
  __shared__ bf16_t s3[64 * 64];
  int b = blockIdx.x, tid = threadIdx.x;
  int lane = tid & 31, w = tid >> 5;
  __builtin_prefetch(W, 0, 0);
  stageT<128>(x1 + b * 4096, s1, tid);
  stageT<128>(x2 + b * 4096, s2, tid);
  stageT<128>(x3 + b * 4096, s3, tid);
  __syncthreads();

  int mt = w;  // 4 waves, 4 M tiles
#pragma unroll
  for (int nt = 0; nt < 4; ++nt) {
    v8f c1 = vzero8(), c2 = vzero8(), c3 = vzero8();
#pragma unroll
    for (int ks = 0; ks < 64; ks += 32) {
      v16bf af = gldA(W, 64, mt * 16, ks, lane);
      c1 = wmma_bf16(af, ldsBT(s1, 64, ks, nt * 16, lane), c1);
      c2 = wmma_bf16(af, ldsBT(s2, 64, ks, nt * 16, lane), c2);
      c3 = wmma_bf16(af, ldsBT(s3, 64, ks, nt * 16, lane), c3);
    }
#pragma unroll
    for (int r = 0; r < 8; ++r) {
      int o = mt * 16 + r + 8 * (lane >> 4);
      int s = nt * 16 + (lane & 15);
      float bias = cb[o];
      float v1 = c1[r] + bias, v2 = c2[r] + bias, v3 = c3[r] + bias;
      int idx = b * 4096 + o * 64 + s;
      A1[idx] = v1; A2[idx] = v1 * v2; A3[idx] = v1 * v3;
    }
  }
}

// ---------------------------------------------------------------------------
// Kernel 2: full Mamba; one block per batch, 8 waves.
// LDS: sX 8K + sU 16K + sZ 16K + sDT 16K + sDTr .5K + sBm 2K + sCm 2K = 60.5K
// ---------------------------------------------------------------------------
__global__ __launch_bounds__(256) void k_mamba(
    const float* __restrict__ Xin,   // (B,64,64) input (also residual)
    const float* __restrict__ Wip,   // (256,64)  in_proj_w
    const float* __restrict__ cw,    // (128,4)   conv1d_w
    const float* __restrict__ cbv,   // (128,)    conv1d_b
    const float* __restrict__ Xpw,   // (36,128)  x_proj_w
    const float* __restrict__ dtW,   // (128,4)   dt_proj_w
    const float* __restrict__ dtB,   // (128,)    dt_proj_b
    const float* __restrict__ Alog,  // (128,16)
    const float* __restrict__ Dvec,  // (128,)
    const float* __restrict__ Wop,   // (64,128)  out_proj_w
    float* __restrict__ Mout) {
  __shared__ bf16_t sX[64 * 64];    // input, row-major (A of GEMM1)
  __shared__ bf16_t sU[64 * 128];   // xi -> u
  __shared__ bf16_t sZ[64 * 128];   // silu(z) -> yssm
  __shared__ bf16_t sDT[64 * 128];  // dt
  __shared__ bf16_t sDTr[64 * 4];   // dt_rank part of x_dbl
  __shared__ bf16_t sBm[64 * 16];   // B rows of x_dbl
  __shared__ bf16_t sCm[64 * 16];   // C rows of x_dbl

  int b = blockIdx.x, tid = threadIdx.x;
  int lane = tid & 31, w = tid >> 5;
  const float* xb = Xin + b * 4096;
  __builtin_prefetch(Wip, 0, 0);
  __builtin_prefetch(Xpw, 0, 0);
  __builtin_prefetch(Wop, 0, 0);

  // Phase 1: stage input as bf16 (row-major), vectorized b64 stores
  for (int i4 = tid; i4 < 1024; i4 += 256) {
    float4 v = ((const float4*)xb)[i4];
    v4bf o = {(bf16_t)v.x, (bf16_t)v.y, (bf16_t)v.z, (bf16_t)v.w};
    *(v4bf*)(sX + i4 * 4) = o;
  }
  __syncthreads();

  // Phase 2: GEMM1 in_proj (64x64)x(64x256); 64 tiles, 8 per wave
#pragma unroll
  for (int j = 0; j < 8; ++j) {
    int job = w * 8 + j;
    int mt = job >> 4, nt = job & 15;
    v8f acc = vzero8();
#pragma unroll
    for (int ks = 0; ks < 64; ks += 32) {
      acc = wmma_bf16(ldsA(sX, 64, mt * 16, ks, lane),
                      gldB(Wip, 64, nt * 16, ks, lane, 256), acc);
    }
#pragma unroll
    for (int r = 0; r < 8; ++r) {
      int L = mt * 16 + r + 8 * (lane >> 4);
      int e = nt * 16 + (lane & 15);
      float v = acc[r];
      if (e < 128) sU[L * 128 + e] = (bf16_t)v;                        // xi
      else         sZ[L * 128 + (e - 128)] = (bf16_t)(v * sigmoidf_(v)); // silu(z)
    }
  }
  __syncthreads();

  // Phase 3: depthwise conv1d (k=4, left pad 3) + silu; 8-channel vectors.
  // 1024 jobs: job -> (l = job>>4, chan group g = job&15), 4 jobs/thread.
  float acc3[4][8];
#pragma unroll
  for (int t = 0; t < 4; ++t) {
    int job = tid + t * 256;
    int l = job >> 4, g = job & 15;
#pragma unroll
    for (int j = 0; j < 8; ++j) acc3[t][j] = cbv[g * 8 + j];
#pragma unroll
    for (int k = 0; k < 4; ++k) {
      int ll = l - 3 + k;
      if (ll >= 0) {
        v8bf xv = *(const v8bf*)(sU + ll * 128 + g * 8);
#pragma unroll
        for (int j = 0; j < 8; ++j)
          acc3[t][j] = fmaf(cw[(g * 8 + j) * 4 + k], (float)xv[j], acc3[t][j]);
      }
    }
#pragma unroll
    for (int j = 0; j < 8; ++j) {
      float y = acc3[t][j];
      acc3[t][j] = y * sigmoidf_(y);  // silu
    }
  }
  __syncthreads();
#pragma unroll
  for (int t = 0; t < 4; ++t) {
    int job = tid + t * 256;
    int l = job >> 4, g = job & 15;
    v8bf o;
#pragma unroll
    for (int j = 0; j < 8; ++j) o[j] = (bf16_t)acc3[t][j];
    *(v8bf*)(sU + l * 128 + g * 8) = o;
  }
  __syncthreads();

  // Phase 4: GEMM2 x_proj (64x128)x(128x36->48); 12 tiles
  for (int jj = w; jj < 12; jj += 8) {
    int mt = jj / 3, nt = jj % 3;
    v8f acc = vzero8();
#pragma unroll
    for (int ks = 0; ks < 128; ks += 32) {
      acc = wmma_bf16(ldsA(sU, 128, mt * 16, ks, lane),
                      gldB(Xpw, 128, nt * 16, ks, lane, 36), acc);
    }
#pragma unroll
    for (int r = 0; r < 8; ++r) {
      int L = mt * 16 + r + 8 * (lane >> 4);
      int e = nt * 16 + (lane & 15);
      bf16_t v = (bf16_t)acc[r];
      if (e < 4)        sDTr[L * 4 + e] = v;
      else if (e < 20)  sBm[L * 16 + (e - 4)] = v;
      else if (e < 36)  sCm[L * 16 + (e - 20)] = v;
    }
  }
  __syncthreads();

  // Phase 5: dt = softplus(dt_r @ dt_proj_w^T + b); 8-channel vector stores
#pragma unroll
  for (int t = 0; t < 4; ++t) {
    int job = tid + t * 256;
    int l = job >> 4, g = job & 15;
    float r0 = (float)sDTr[l * 4 + 0], r1 = (float)sDTr[l * 4 + 1];
    float r2 = (float)sDTr[l * 4 + 2], r3 = (float)sDTr[l * 4 + 3];
    v8bf o;
#pragma unroll
    for (int j = 0; j < 8; ++j) {
      int d = g * 8 + j;
      float v = dtB[d];
      v = fmaf(r0, dtW[d * 4 + 0], v);
      v = fmaf(r1, dtW[d * 4 + 1], v);
      v = fmaf(r2, dtW[d * 4 + 2], v);
      v = fmaf(r3, dtW[d * 4 + 3], v);
      o[j] = (bf16_t)softplusf_(v);
    }
    *(v8bf*)(sDT + l * 128 + g * 8) = o;
  }
  __syncthreads();

  // Phase 6: selective scan (sequential over L); 128 threads, 16 states each
  if (tid < 128) {
    int ch = tid;
    float Ar[16], h[16];
#pragma unroll
    for (int n = 0; n < 16; ++n) {
      Ar[n] = -__expf(Alog[ch * 16 + n]);
      h[n] = 0.f;
    }
    float Dv = Dvec[ch];
    for (int l = 0; l < 64; ++l) {
      float dt = (float)sDT[l * 128 + ch];
      float u  = (float)sU[l * 128 + ch];
      float dtu = dt * u;
      v16bf Bv = cat8(((const v8bf*)(sBm + l * 16))[0],
                      ((const v8bf*)(sBm + l * 16))[1]);
      v16bf Cv = cat8(((const v8bf*)(sCm + l * 16))[0],
                      ((const v8bf*)(sCm + l * 16))[1]);
      float y = 0.f;
#pragma unroll
      for (int n = 0; n < 16; ++n) {
        float dA = __expf(dt * Ar[n]);
        h[n] = fmaf(dA, h[n], dtu * (float)Bv[n]);
        y = fmaf(h[n], (float)Cv[n], y);
      }
      float yv = y + u * Dv;
      float zz = (float)sZ[l * 128 + ch];       // silu(z)
      sZ[l * 128 + ch] = (bf16_t)(yv * zz);     // yssm in place
    }
  }
  __syncthreads();

  // Phase 7: GEMM3 out_proj (64x128)x(128x64) + residual; 16 tiles
#pragma unroll
  for (int j = 0; j < 2; ++j) {
    int job = w * 2 + j;
    int mt = job >> 2, nt = job & 3;
    v8f acc = vzero8();
#pragma unroll
    for (int ks = 0; ks < 128; ks += 32) {
      acc = wmma_bf16(ldsA(sZ, 128, mt * 16, ks, lane),
                      gldB(Wop, 128, nt * 16, ks, lane, 64), acc);
    }
#pragma unroll
    for (int r = 0; r < 8; ++r) {
      int row = mt * 16 + r + 8 * (lane >> 4);
      int col = nt * 16 + (lane & 15);
      int idx = b * 4096 + row * 64 + col;
      Mout[idx] = acc[r] + xb[row * 64 + col];  // residual
    }
  }
}

// ---------------------------------------------------------------------------
// Kernel 3: zero BN accumulators (d_ws is poisoned; must init every call)
// ---------------------------------------------------------------------------
__global__ void k_zero(float* __restrict__ stats) {
  int i = threadIdx.x;
  if (i < 384) stats[i] = 0.f;
}

// ---------------------------------------------------------------------------
// Kernel 4: conv1x1 on each mamba output + per-channel sum/sumsq for BN
// ---------------------------------------------------------------------------
__global__ __launch_bounds__(128) void k_conv_bn(
    const float* __restrict__ Mall, const float* __restrict__ W,
    const float* __restrict__ cb, float* __restrict__ Call,
    float* __restrict__ stats) {
  __shared__ bf16_t sBt[64 * 64];   // transposed activation
  __shared__ float lsum[64];
  __shared__ float lsq[64];
  int b = blockIdx.x, t = blockIdx.y;
  int tid = threadIdx.x, lane = tid & 31, w = tid >> 5;
  const float* src = Mall + (size_t)t * NTENS_ELEMS + b * 4096;
  float* dst = Call + (size_t)t * NTENS_ELEMS + b * 4096;
  float* gs = stats + t * 128;
  __builtin_prefetch(W, 0, 0);

  stageT<128>(src, sBt, tid);
  if (tid < 64) { lsum[tid] = 0.f; lsq[tid] = 0.f; }
  __syncthreads();

  int mt = w;
#pragma unroll
  for (int nt = 0; nt < 4; ++nt) {
    v8f acc = vzero8();
#pragma unroll
    for (int ks = 0; ks < 64; ks += 32) {
      acc = wmma_bf16(gldA(W, 64, mt * 16, ks, lane),
                      ldsBT(sBt, 64, ks, nt * 16, lane), acc);
    }
#pragma unroll
    for (int r = 0; r < 8; ++r) {
      int o = mt * 16 + r + 8 * (lane >> 4);
      int s = nt * 16 + (lane & 15);
      float v = acc[r] + cb[o];
      dst[o * 64 + s] = v;
      atomicAdd(&lsum[o], v);        // ds_add_f32
      atomicAdd(&lsq[o], v * v);
    }
  }
  __syncthreads();
  if (tid < 64) {
    atomicAdd(&gs[tid], lsum[tid]);      // global_atomic_add_f32
    atomicAdd(&gs[64 + tid], lsq[tid]);
  }
}

// ---------------------------------------------------------------------------
// Kernel 5: batchnorm + relu + weighted combine -> d_out
// ---------------------------------------------------------------------------
__global__ __launch_bounds__(256) void k_final(
    const float* __restrict__ Call, const float* __restrict__ stats,
    const float* __restrict__ gamma, const float* __restrict__ beta,
    const float* __restrict__ xs1, const float* __restrict__ xs2,
    float* __restrict__ out) {
  int idx = blockIdx.x * 256 + threadIdx.x;
  if (idx >= NTENS_ELEMS) return;
  int o = (idx >> 6) & 63;
  const float inv = 1.f / 16384.f;  // BATCH * SEQ
  float r[3];
#pragma unroll
  for (int t = 0; t < 3; ++t) {
    float v = Call[(size_t)t * NTENS_ELEMS + idx];
    float mean = stats[t * 128 + o] * inv;
    float var = stats[t * 128 + 64 + o] * inv - mean * mean;
    float rr = gamma[o] * (v - mean) * rsqrtf(var + 1e-5f) + beta[o];
    r[t] = fmaxf(rr, 0.f);
  }
  out[idx] = r[0] + xs1[0] * r[1] + xs2[0] * r[2];
}

// ---------------------------------------------------------------------------
extern "C" void kernel_launch(void* const* d_in, const int* in_sizes, int n_in,
                              void* d_out, int out_size, void* d_ws, size_t ws_size,
                              hipStream_t stream) {
  (void)in_sizes; (void)n_in; (void)out_size; (void)ws_size;
  const float* x1        = (const float*)d_in[0];
  const float* x2        = (const float*)d_in[1];
  const float* x3        = (const float*)d_in[2];
  const float* conv1_w   = (const float*)d_in[3];
  const float* conv1_b   = (const float*)d_in[4];
  const float* bn_gamma  = (const float*)d_in[5];
  const float* bn_beta   = (const float*)d_in[6];
  const float* xishu1    = (const float*)d_in[7];
  const float* xishu2    = (const float*)d_in[8];
  const float* in_proj_w = (const float*)d_in[9];
  const float* conv1d_w  = (const float*)d_in[10];
  const float* conv1d_b  = (const float*)d_in[11];
  const float* x_proj_w  = (const float*)d_in[12];
  const float* dt_proj_w = (const float*)d_in[13];
  const float* dt_proj_b = (const float*)d_in[14];
  const float* A_log     = (const float*)d_in[15];
  const float* Dvec      = (const float*)d_in[16];
  const float* out_proj_w= (const float*)d_in[17];

  // Workspace layout (floats): A[3], M[3], C[3] tensors + 384 stats (~38 MB)
  float* ws    = (float*)d_ws;
  float* Abuf  = ws;
  float* Mbuf  = ws + (size_t)3 * NTENS_ELEMS;
  float* Cbuf  = ws + (size_t)6 * NTENS_ELEMS;
  float* stats = ws + (size_t)9 * NTENS_ELEMS;

  k_front_conv<<<BATCH, 128, 0, stream>>>(x1, x2, x3, conv1_w, conv1_b,
                                          Abuf, Abuf + NTENS_ELEMS,
                                          Abuf + 2 * (size_t)NTENS_ELEMS);

  for (int t = 0; t < 3; ++t) {
    k_mamba<<<BATCH, 256, 0, stream>>>(
        Abuf + (size_t)t * NTENS_ELEMS, in_proj_w, conv1d_w, conv1d_b,
        x_proj_w, dt_proj_w, dt_proj_b, A_log, Dvec, out_proj_w,
        Mbuf + (size_t)t * NTENS_ELEMS);
  }

  k_zero<<<1, 384, 0, stream>>>(stats);
  k_conv_bn<<<dim3(BATCH, 3), 128, 0, stream>>>(Mbuf, conv1_w, conv1_b,
                                                Cbuf, stats);
  k_final<<<NTENS_ELEMS / 256, 256, 0, stream>>>(Cbuf, stats, bn_gamma,
                                                 bn_beta, xishu1, xishu2,
                                                 (float*)d_out);
}
